// Pe_global_based_attention_variation_27367531610179
// MI455X (gfx1250) — matrix operand order repaired and checked by reading, not compiled, not run on hardware
//
#include <hip/hip_runtime.h>
#include <math.h>

// Pe_global_based_attention (L2-distance attention), MI455X / gfx1250, wave32.
// S_ij = (2*q.k - kk_j)/sqrt(D)  (qq_i cancels in softmax); flash-attention
// online softmax in "transposed" orientation so per-query softmax stats live
// in a lane pair (lane, lane^16).  Global->LDS staging is software-pipelined
// through registers so global_load latency overlaps the WMMA/softmax phase.

typedef __attribute__((ext_vector_type(16))) _Float16 v16h;
typedef __attribute__((ext_vector_type(8)))  _Float16 v8h;
typedef __attribute__((ext_vector_type(8)))  float    v8f;
typedef __attribute__((ext_vector_type(4)))  float    v4f;

#define B_ 4
#define H_ 8
#define N_ 2048
#define D_ 64

constexpr int QTILE = 128;          // queries per workgroup (8 waves x 16)
constexpr int KTILE = 32;           // keys per main-loop iteration
constexpr float INV_SCALE = 0.125f; // 1/sqrt(64)

// Build a 16x32 f16 A-frag / 32x16 f16 B-frag lane slice from two contiguous
// 8-half (16B) LDS chunks -> two ds_load_b128.
__device__ inline v16h make_frag(const _Float16* lo, const _Float16* hi) {
  v8h a = *(const v8h*)lo;
  v8h b = *(const v8h*)hi;
  return __builtin_shufflevector(a, b, 0, 1, 2, 3, 4, 5, 6, 7,
                                       8, 9, 10, 11, 12, 13, 14, 15);
}

// Register stage for one 32x64 K tile + one 32x64 V tile (per-thread slice:
// 2 float4 of K, 2 float4 of V).
struct Stage {
  v4f k0, k1, v0, v1;
};

__device__ inline Stage stage_load(const float* __restrict__ kbase,
                                   const float* __restrict__ vbase, int tid) {
  Stage s;
  s.k0 = *(const v4f*)(kbase + (size_t)tid * 4);
  s.k1 = *(const v4f*)(kbase + (size_t)(tid + 256) * 4);
  s.v0 = *(const v4f*)(vbase + (size_t)tid * 4);
  s.v1 = *(const v4f*)(vbase + (size_t)(tid + 256) * 4);
  return s;
}

__global__ __launch_bounds__(256)
void pe_l2_attention_kernel(const float* __restrict__ Qg,
                            const float* __restrict__ Kg,
                            const float* __restrict__ Vg,
                            float* __restrict__ Og) {
  __shared__ __align__(16) _Float16 sQ[QTILE * D_];   // 16 KB, row-major
  __shared__ __align__(16) _Float16 sK[KTILE * D_];   // 4 KB, row-major
  __shared__ __align__(16) _Float16 sVT[D_ * KTILE];  // 4 KB, transposed (d x key)
  __shared__ __align__(16) float    sKK[KTILE];       // per-key |k|^2

  const int nQT = N_ / QTILE;
  const int bh  = blockIdx.x / nQT;        // fused (b,h)
  const int qt  = blockIdx.x % nQT;
  const int q0  = qt * QTILE;
  const int b   = bh / H_;
  const int h   = bh % H_;

  const int tid  = threadIdx.x;
  const int lane = tid & 31;
  const int wave = tid >> 5;
  const int lh   = lane & 15;              // position within half-wave
  const int hi   = lane >> 4;              // 0: lanes 0-15, 1: lanes 16-31

  const size_t headOff = (size_t)bh * N_ * D_;
  const float* qg = Qg + headOff + (size_t)q0 * D_;
  const float* kg = Kg + headOff;
  const float* vg = Vg + headOff;

  // ---------------- load Q tile: f32 -> f16, row major ----------------
  {
    const int nVec = QTILE * D_ / 4;       // 2048 float4 chunks
    for (int i = tid; i < nVec; i += 256) {
      v4f f = *(const v4f*)(qg + (size_t)i * 4);
      #pragma unroll
      for (int e = 0; e < 4; ++e) sQ[i * 4 + e] = (_Float16)f[e];
    }
  }

  // Prefetch first K/V tile into registers (overlaps with Q staging above).
  Stage cur = stage_load(kg, vg, tid);

  __syncthreads();

  // Per-wave Q^T B-fragments (N = 16 queries of this wave, K-dim = d).
  // B layout: lanes 0-15 hold column N=lane with K=0..7 & 16..23, lanes 16-31
  // hold the same columns with K=8..15 & 24..31.
  const int qrow = wave * 16 + lh;
  const _Float16* qr = sQ + qrow * D_;
  const v16h bq0 = make_frag(qr + hi * 8,      qr + 16 + hi * 8);   // d 0..31
  const v16h bq1 = make_frag(qr + 32 + hi * 8, qr + 48 + hi * 8);  // d 32..63

  v8f acc[4] = {};                 // O^T accumulators: d-tiles 0..3, q = lane&15
  float m = -INFINITY;             // running max  (per query, dup lane/lane^16)
  float l = 0.0f;                  // running sum

  // Per-thread vec4 chunk mapping within a tile (i = tid, tid+256):
  //   chunk i covers key row i/16, d columns (i%16)*4 .. +3.
  const int row0 = tid >> 4;                 // = 2*wave + hi
  const int col0 = (tid & 15) * 4;

  for (int j0 = 0; j0 < N_; j0 += KTILE) {
    // ------------- store staged K/V tile to LDS (f32 -> f16) -------------
    {
      #pragma unroll
      for (int e = 0; e < 4; ++e) {
        sK[row0 * D_ + col0 + e]        = (_Float16)cur.k0[e];
        sK[(row0 + 16) * D_ + col0 + e] = (_Float16)cur.k1[e];
        sVT[(col0 + e) * KTILE + row0]        = (_Float16)cur.v0[e];
        sVT[(col0 + e) * KTILE + (row0 + 16)] = (_Float16)cur.v1[e];
      }
      // kk_j from staged registers: sum of squares reduced over the 16
      // threads (one 16-lane half-wave group) that share a key row.
      float ss0 = cur.k0[0] * cur.k0[0] + cur.k0[1] * cur.k0[1] +
                  cur.k0[2] * cur.k0[2] + cur.k0[3] * cur.k0[3];
      float ss1 = cur.k1[0] * cur.k1[0] + cur.k1[1] * cur.k1[1] +
                  cur.k1[2] * cur.k1[2] + cur.k1[3] * cur.k1[3];
      #pragma unroll
      for (int off = 1; off < 16; off <<= 1) {
        ss0 += __shfl_xor(ss0, off, 32);
        ss1 += __shfl_xor(ss1, off, 32);
      }
      if (lh == 0) {                         // lanes 0 and 16 of each wave
        sKK[row0]      = ss0;                // rows 2w, 2w+1
        sKK[row0 + 16] = ss1;                // rows 16+2w, 16+2w+1
      }
    }
    __syncthreads();

    // Prefetch next tile: loads issue now, results consumed only after the
    // end-of-iteration barrier -> latency hidden behind WMMA + softmax.
    Stage nxt = cur;
    if (j0 + KTILE < N_)
      nxt = stage_load(kg + (size_t)(j0 + KTILE) * D_,
                       vg + (size_t)(j0 + KTILE) * D_, tid);

    // ------------- S^T = K . Q^T : two 16(key)x16(query) tiles -------------
    // A layout (16x32 f16): lanes 0-15: M=lane, K=0..7 & 16..23;
    //                       lanes 16-31: M=lane-16, K=8..15 & 24..31.
    const _Float16* k0 = sK + lh * D_;          // key = lh
    const _Float16* k1 = sK + (16 + lh) * D_;   // key = 16+lh
    v8f c0 = {}, c1 = {};
    v16h a;
    a  = make_frag(k0 + hi * 8,      k0 + 16 + hi * 8);        // d 0..31
    c0 = __builtin_amdgcn_wmma_f32_16x16x32_f16(false, a, false, bq0,
                                                (short)0, c0, false, false);
    a  = make_frag(k0 + 32 + hi * 8, k0 + 48 + hi * 8);        // d 32..63
    c0 = __builtin_amdgcn_wmma_f32_16x16x32_f16(false, a, false, bq1,
                                                (short)0, c0, false, false);
    a  = make_frag(k1 + hi * 8,      k1 + 16 + hi * 8);
    c1 = __builtin_amdgcn_wmma_f32_16x16x32_f16(false, a, false, bq0,
                                                (short)0, c1, false, false);
    a  = make_frag(k1 + 32 + hi * 8, k1 + 48 + hi * 8);
    c1 = __builtin_amdgcn_wmma_f32_16x16x32_f16(false, a, false, bq1,
                                                (short)0, c1, false, false);

    // ------------- scores + online softmax (per query = per lane pair) -----
    const v4f kkA0 = *(const v4f*)(sKK + hi * 8);
    const v4f kkA1 = *(const v4f*)(sKK + hi * 8 + 4);
    const v4f kkB0 = *(const v4f*)(sKK + 16 + hi * 8);
    const v4f kkB1 = *(const v4f*)(sKK + 16 + hi * 8 + 4);

    float s0[8], s1[8];
    float bm = -INFINITY;
    #pragma unroll
    for (int r = 0; r < 8; ++r) {
      const float kkA = (r < 4) ? kkA0[r & 3] : kkA1[r & 3];
      const float kkB = (r < 4) ? kkB0[r & 3] : kkB1[r & 3];
      s0[r] = (2.0f * c0[r] - kkA) * INV_SCALE;   // keys hi*8 + r
      s1[r] = (2.0f * c1[r] - kkB) * INV_SCALE;   // keys 16 + hi*8 + r
      bm = fmaxf(bm, fmaxf(s0[r], s1[r]));
    }
    bm = fmaxf(bm, __shfl_xor(bm, 16, 32));       // combine lane pair
    const float mNew  = fmaxf(m, bm);
    const float alpha = __expf(m - mNew);
    m = mNew;

    float rs = 0.0f;
    _Float16 p0[8], p1[8];
    #pragma unroll
    for (int r = 0; r < 8; ++r) {
      const float e0 = __expf(s0[r] - mNew);
      const float e1 = __expf(s1[r] - mNew);
      rs += e0 + e1;
      p0[r] = (_Float16)e0;
      p1[r] = (_Float16)e1;
    }
    rs += __shfl_xor(rs, 16, 32);
    l = l * alpha + rs;

    #pragma unroll
    for (int t = 0; t < 4; ++t)
      #pragma unroll
      for (int r = 0; r < 8; ++r) acc[t][r] *= alpha;

    // ------------- build P^T (32 keys x 16 queries) B-fragment -------------
    // Lane<16 needs keys 0..15 of its query; lane>=16 needs keys 16..31.
    // Own C tiles supply half of each; the other half comes from lane^16.
    unsigned pk0[4], pk1[4];
    #pragma unroll
    for (int j = 0; j < 4; ++j) {
      union { _Float16 hh[2]; unsigned u; } u0, u1;
      u0.hh[0] = p0[2 * j]; u0.hh[1] = p0[2 * j + 1];
      u1.hh[0] = p1[2 * j]; u1.hh[1] = p1[2 * j + 1];
      pk0[j] = u0.u; pk1[j] = u1.u;
    }
    union { unsigned u[8]; v16h v; } bp;
    #pragma unroll
    for (int j = 0; j < 4; ++j) {
      const unsigned o0 = (unsigned)__shfl_xor((int)pk0[j], 16, 32);
      const unsigned o1 = (unsigned)__shfl_xor((int)pk1[j], 16, 32);
      bp.u[j]     = hi ? o1 : pk0[j];   // halves 0..7  : keys 0..7  | 16..23
      bp.u[4 + j] = hi ? pk1[j] : o0;   // halves 8..15 : keys 8..15 | 24..31
    }

    // ------------- O^T += V^T . P^T  (4 d-tiles, K-dim = 32 keys) ----------
    #pragma unroll
    for (int t = 0; t < 4; ++t) {
      const _Float16* vr = sVT + (t * 16 + lh) * KTILE;   // row d = 16t + lh
      const v16h av = make_frag(vr + hi * 16, vr + hi * 16 + 8);
      acc[t] = __builtin_amdgcn_wmma_f32_16x16x32_f16(false, av, false, bp.v,
                                                      (short)0, acc[t],
                                                      false, false);
    }
    __syncthreads();   // safe to overwrite sK/sVT/sKK next iteration
    cur = nxt;
  }

  // ---------------- epilogue: out[b, q, h, d] = O^T / l ----------------
  const float invl = 1.0f / l;
  float* orow = Og + (((size_t)b * N_ + (size_t)(q0 + wave * 16 + lh)) * H_ + h) * D_;
  #pragma unroll
  for (int t = 0; t < 4; ++t) {
    v4f w0, w1;
    #pragma unroll
    for (int e = 0; e < 4; ++e) {
      w0[e] = acc[t][e] * invl;
      w1[e] = acc[t][4 + e] * invl;
    }
    const int d0 = t * 16 + hi * 8;       // lanes 0-15: M=r, lanes 16-31: M=8+r
    *(v4f*)(orow + d0)     = w0;
    *(v4f*)(orow + d0 + 4) = w1;
  }
}

extern "C" void kernel_launch(void* const* d_in, const int* in_sizes, int n_in,
                              void* d_out, int out_size, void* d_ws, size_t ws_size,
                              hipStream_t stream) {
  (void)in_sizes; (void)n_in; (void)d_ws; (void)ws_size; (void)out_size;
  const float* q = (const float*)d_in[0];
  const float* k = (const float*)d_in[1];
  const float* v = (const float*)d_in[2];
  float* o = (float*)d_out;
  dim3 grid(B_ * H_ * (N_ / QTILE));   // 512 workgroups
  pe_l2_attention_kernel<<<grid, 256, 0, stream>>>(q, k, v, o);
}
